// Attention_24507083391089
// MI455X (gfx1250) — matrix operand order repaired and checked by reading, not compiled, not run on hardware
//
#include <hip/hip_runtime.h>
#include <stdint.h>

// ---------------- problem constants ----------------
#define D_MODEL   1024
#define N_HEADS   16
#define D_HEAD    64
#define SEQ       2048
#define BATCH     4
#define MROWS     (BATCH * SEQ)      // 8192 total token rows
#define IGNORE_VAL -100000.0f
#define SCALE      0.125f            // 1/sqrt(64)

// ---------------- WMMA plumbing ----------------
typedef __attribute__((ext_vector_type(16))) __bf16 v16bf;
typedef __attribute__((ext_vector_type(8)))  __bf16 v8bf;    // one b128
typedef __attribute__((ext_vector_type(2)))  __bf16 v2bf;    // one b32
typedef __attribute__((ext_vector_type(8)))  float  v8f;

typedef __attribute__((ext_vector_type(4))) uint32_t v4u;
typedef __attribute__((ext_vector_type(8))) int      v8i_t;
typedef __attribute__((ext_vector_type(4))) int      v4i_t;

#if defined(__has_builtin)
#  if __has_builtin(__builtin_amdgcn_tensor_load_to_lds)
#    define HAVE_TDM 1
#  endif
#endif

__device__ __forceinline__ v8f wmma_bf16(v16bf a, v16bf b, v8f c) {
  return __builtin_amdgcn_wmma_f32_16x16x32_bf16(false, a, false, b,
                                                 (short)0, c, false, false);
}

__device__ __forceinline__ v8f zero8() {
  v8f r;
#pragma unroll
  for (int i = 0; i < 8; ++i) r[i] = 0.0f;
  return r;
}

// Fragment = two 16-byte chunks (ds_load_b128 / global_load_b128 each).
__device__ __forceinline__ v16bf frag_cat(const __bf16* p0, const __bf16* p1) {
  v8bf lo = *(const v8bf*)p0;
  v8bf hi = *(const v8bf*)p1;
  return __builtin_shufflevector(lo, hi, 0, 1, 2, 3, 4, 5, 6, 7,
                                         8, 9, 10, 11, 12, 13, 14, 15);
}

// pack two f32 -> one b32 bf16 pair (fuses to v_cvt_pk_bf16_f32)
__device__ __forceinline__ v2bf pack2(float a, float b) {
  v2bf r; r[0] = (__bf16)a; r[1] = (__bf16)b; return r;
}

// reductions across the 16-lane halves of a wave32 (rows of the C layout)
__device__ __forceinline__ float half16_max(float v) {
  v = fmaxf(v, __shfl_xor(v, 1, 32));
  v = fmaxf(v, __shfl_xor(v, 2, 32));
  v = fmaxf(v, __shfl_xor(v, 4, 32));
  v = fmaxf(v, __shfl_xor(v, 8, 32));
  return v;
}
__device__ __forceinline__ float half16_sum(float v) {
  v += __shfl_xor(v, 1, 32);
  v += __shfl_xor(v, 2, 32);
  v += __shfl_xor(v, 4, 32);
  v += __shfl_xor(v, 8, 32);
  return v;
}

__device__ __forceinline__ void wait_tensorcnt0() {
#if defined(__has_builtin)
#  if __has_builtin(__builtin_amdgcn_s_wait_tensorcnt)
  __builtin_amdgcn_s_wait_tensorcnt(0);
#  else
  asm volatile("s_wait_tensorcnt 0x0" ::: "memory");
#  endif
#else
  asm volatile("s_wait_tensorcnt 0x0" ::: "memory");
#endif
}

#if defined(HAVE_TDM)
// 2D TDM load: 64 rows x 64 bf16, global row stride 1024 elements, into
// LDS at `lds_byte` with +16B pad per 128B row ([64][64+8] bf16 pitch).
__device__ __forceinline__ void tdm_load_tile_bf16(const __bf16* gptr,
                                                   uint32_t lds_byte) {
  uint64_t ga = (uint64_t)(uintptr_t)gptr;
  v4u g0;
  g0[0] = 1u;                                   // count=1, user descriptor
  g0[1] = lds_byte;                             // lds_addr
  g0[2] = (uint32_t)ga;                         // global_addr[31:0]
  g0[3] = (uint32_t)((ga >> 32) & 0x01FFFFFFu)  // global_addr[56:32]
          | 0x80000000u;                        // type=2 ("image")
  v8i_t g1;
  g1[0] = (int)((1u << 16)     // data_size = 2 bytes (bf16)
              | (1u << 20)     // pad_enable
              | (4u << 22)     // pad_interval code 4 = every 32 DWORDs
              | (3u << 25));   // pad_amount  code 3 = 4 DWORDs (16B)
  g1[1] = (int)(64u << 16);                     // tensor_dim0 = 64 (lo16)
  g1[2] = (int)(64u << 16);                     // dim0 hi=0 | tensor_dim1=64
  g1[3] = (int)(64u << 16);                     // dim1 hi=0 | tile_dim0=64
  g1[4] = (int)64;                              // tile_dim1=64, tile_dim2=0
  g1[5] = (int)D_MODEL;                         // tensor_dim0_stride = 1024
  g1[6] = 0;
  g1[7] = 0;
  v4i_t gz;
  gz[0] = 0; gz[1] = 0; gz[2] = 0; gz[3] = 0;
#  if __has_include(<hip/amd_detail/amd_gfx1250_TDM.h>)
  v8i_t z8;
#pragma unroll
  for (int i = 0; i < 8; ++i) z8[i] = 0;
  __builtin_amdgcn_tensor_load_to_lds(g0, g1, gz, gz, z8, 0);
#  else
  __builtin_amdgcn_tensor_load_to_lds(g0, g1, gz, gz, 0);
#  endif
}
#endif

// =====================================================================
// Kernel 1: per-head QKV projection.  X[8192,1024] (fp32) x W[n][1024][64]
// -> Out bf16 [8192][1024]  (col = head*64 + dh  == bsnh flattened)
// =====================================================================
__global__ __launch_bounds__(256)
void qkv_proj_kernel(const float* __restrict__ X,
                     const float* __restrict__ W,     // [16][1024][64]
                     const float* __restrict__ bias,  // [1024]
                     __bf16* __restrict__ Out) {
  __shared__ __bf16 As[128][32 + 8];    // 80B pitch
  __shared__ __bf16 BsT[64][32 + 8];    // transposed: [col][k]

  const int tid  = threadIdx.x;
  const int wid  = tid >> 5;
  const int lane = tid & 31;
  const int r0   = blockIdx.x * 128;
  const int c0   = blockIdx.y * 64;
  const int head = blockIdx.y;

  v8f acc[4];
#pragma unroll
  for (int i = 0; i < 4; ++i) acc[i] = zero8();

  const int arow = wid * 16 + (lane & 15);
  const int kbA  = (lane < 16) ? 0 : 8;
  const int kbB  = (lane < 16) ? 0 : 8;

  for (int kk = 0; kk < D_MODEL; kk += 32) {
    __syncthreads();
#pragma unroll
    for (int i = 0; i < 8; ++i) {             // 128x32 fp32 -> bf16 pairs
      int e = tid + i * 256;                  // 2048 pairs
      int row = e >> 4, cp = (e & 15) * 2;
      const float* xp = &X[(size_t)(r0 + row) * D_MODEL + kk + cp];
      *(v2bf*)&As[row][cp] = pack2(xp[0], xp[1]);
    }
#pragma unroll
    for (int i = 0; i < 4; ++i) {             // 32x64 weights, transposed
      int e = tid + i * 256;                  // 1024 pairs (along k)
      int col = e & 63, krp = (e >> 6) * 2;
      size_t wb = (size_t)head * (D_MODEL * D_HEAD) +
                  (size_t)(kk + krp) * D_HEAD + col;
      *(v2bf*)&BsT[col][krp] = pack2(W[wb], W[wb + D_HEAD]);
    }
    __syncthreads();

    v16bf a = frag_cat(&As[arow][kbA], &As[arow][kbA + 16]);
#pragma unroll
    for (int bn = 0; bn < 4; ++bn) {
      int col = bn * 16 + (lane & 15);
      v16bf bfrag = frag_cat(&BsT[col][kbB], &BsT[col][kbB] + 16);
      acc[bn] = wmma_bf16(a, bfrag, acc[bn]);
    }
  }

  const int mof = (lane < 16) ? 0 : 8;
#pragma unroll
  for (int bn = 0; bn < 4; ++bn) {
    int col = c0 + bn * 16 + (lane & 15);
    float bv = bias[col];
#pragma unroll
    for (int i = 0; i < 8; ++i) {
      int row = r0 + wid * 16 + i + mof;
      Out[(size_t)row * D_MODEL + col] = (__bf16)(acc[bn][i] + bv);
    }
  }
}

// =====================================================================
// Kernel 2: flash attention per (batch, head).
// =====================================================================
__global__ __launch_bounds__(256)
void attn_kernel(const __bf16* __restrict__ Q,      // [8192][1024]
                 const __bf16* __restrict__ K,
                 const __bf16* __restrict__ V,
                 const uint8_t* __restrict__ mask,  // [2048][2048] bool
                 float* __restrict__ Zout,
                 __bf16* __restrict__ Zbf) {
  __shared__ __bf16 Kt[64][64 + 8];        // [key][dd]   144B pitch
  __shared__ __bf16 VtT[64][64 + 8];       // [dd][key]   (transposed V)
  __shared__ __bf16 Pt[8][16][64 + 8];     // wave-private P

  const int tid  = threadIdx.x;
  const int wid  = tid >> 5;
  const int lane = tid & 31;
  const int b    = blockIdx.z;
  const int h    = blockIdx.y;
  const int sq0  = blockIdx.x * 128 + wid * 16;
  const size_t rowbase = (size_t)b * SEQ;
  const size_t headoff = (size_t)h * D_HEAD;

  const int arow = lane & 15;
  const int kbA  = (lane < 16) ? 0 : 8;
  const int kbB  = (lane < 16) ? 0 : 8;
  const int mof  = (lane < 16) ? 0 : 8;

  // Q A-fragments: contiguous 16B chunks straight from global
  const __bf16* qrow = Q + (rowbase + sq0 + arow) * D_MODEL + headoff;
  v16bf aq0 = frag_cat(qrow + kbA,      qrow + kbA + 16);
  v16bf aq1 = frag_cat(qrow + 32 + kbA, qrow + 48 + kbA);

  float m[8], ell[8];
#pragma unroll
  for (int i = 0; i < 8; ++i) { m[i] = -3.0e38f; ell[i] = 0.0f; }
  v8f o[4];
#pragma unroll
  for (int i = 0; i < 4; ++i) o[i] = zero8();

  for (int kc = 0; kc < SEQ; kc += 64) {
    __syncthreads();

#if defined(HAVE_TDM)
    if (wid == 0) {                       // one TDM per workgroup chunk
      tdm_load_tile_bf16(K + (rowbase + kc) * D_MODEL + headoff,
                         (uint32_t)(uintptr_t)(&Kt[0][0]));
    }
#endif
#pragma unroll
    for (int i = 0; i < 2; ++i) {         // V tile: b128 reads, transposed
      int e = tid + i * 256;              // 512 groups of 8 elements
      int r = e >> 3, c8 = (e & 7) * 8;
      const __bf16* vp = V + (rowbase + kc + r) * D_MODEL + headoff + c8;
      v8bf vv = *(const v8bf*)vp;
#pragma unroll
      for (int j = 0; j < 8; ++j) VtT[c8 + j][r] = vv[j];
#if !defined(HAVE_TDM)
      const __bf16* kp = K + (rowbase + kc + r) * D_MODEL + headoff + c8;
      *(v8bf*)&Kt[r][c8] = *(const v8bf*)kp;
#endif
    }
    if (kc + 64 < SEQ) {                  // CDNA5 global_prefetch_b8
      size_t gn = (rowbase + kc + 64 + (tid >> 2)) * D_MODEL + headoff +
                  ((size_t)(tid & 3) << 4);
      __builtin_prefetch(&K[gn], 0, 1);
      __builtin_prefetch(&V[gn], 0, 1);
    }
#if defined(HAVE_TDM)
    if (wid == 0) wait_tensorcnt0();
#endif
    __syncthreads();

    // ---- scores for the full 64-key chunk ----
    float sv[4][8];
#pragma unroll
    for (int t4 = 0; t4 < 4; ++t4) {
      int key = t4 * 16 + (lane & 15);
      v16bf bk0 = frag_cat(&Kt[key][kbB],      &Kt[key][kbB] + 16);
      v16bf bk1 = frag_cat(&Kt[key][32 + kbB], &Kt[key][32 + kbB] + 16);
      v8f s = zero8();
      s = wmma_bf16(aq0, bk0, s);
      s = wmma_bf16(aq1, bk1, s);
      int kglob = kc + key;
#pragma unroll
      for (int i = 0; i < 8; ++i) {
        int qglob = sq0 + i + mof;
        float val = s[i] * SCALE;
        if (mask[(size_t)qglob * SEQ + kglob]) val = IGNORE_VAL;
        sv[t4][i] = val;
      }
    }

    // ---- one online-softmax update per chunk ----
#pragma unroll
    for (int i = 0; i < 8; ++i) {
      float mx = fmaxf(fmaxf(sv[0][i], sv[1][i]), fmaxf(sv[2][i], sv[3][i]));
      mx = half16_max(mx);
      float mn   = fmaxf(m[i], mx);
      float corr = __expf(m[i] - mn);
      float psum = 0.0f;
#pragma unroll
      for (int t4 = 0; t4 < 4; ++t4) {
        float p = __expf(sv[t4][i] - mn);
        psum += p;
        Pt[wid][i + mof][t4 * 16 + (lane & 15)] = (__bf16)p;
      }
      ell[i] = ell[i] * corr + half16_sum(psum);
      m[i]   = mn;
#pragma unroll
      for (int nt = 0; nt < 4; ++nt) o[nt][i] *= corr;
    }

    // ---- O += P(16x64) x V(64x64) ----
    const __bf16* prow = &Pt[wid][arow][0];
    v16bf ap0 = frag_cat(prow + kbA,      prow + kbA + 16);
    v16bf ap1 = frag_cat(prow + 32 + kbA, prow + 48 + kbA);
#pragma unroll
    for (int nt = 0; nt < 4; ++nt) {
      int dd = nt * 16 + (lane & 15);
      v16bf bv0 = frag_cat(&VtT[dd][kbB],      &VtT[dd][kbB] + 16);
      v16bf bv1 = frag_cat(&VtT[dd][32 + kbB], &VtT[dd][32 + kbB] + 16);
      o[nt] = wmma_bf16(ap0, bv0, o[nt]);
      o[nt] = wmma_bf16(ap1, bv1, o[nt]);
    }
  }

  // epilogue: z = O / l   (bsnh flattened == row*1024 + h*64 + dd)
#pragma unroll
  for (int nt = 0; nt < 4; ++nt) {
    int dd = nt * 16 + (lane & 15);
#pragma unroll
    for (int i = 0; i < 8; ++i) {
      int q = sq0 + i + mof;
      float val = o[nt][i] / ell[i];
      size_t idx = (rowbase + q) * D_MODEL + headoff + dd;
      Zout[idx] = val;
      Zbf[idx]  = (__bf16)val;
    }
  }
}

// =====================================================================
// Kernel 3: output projection.  z[8192,1024](bf16) x W_O[1024,1024] + b_O
// =====================================================================
__global__ __launch_bounds__(256)
void out_proj_kernel(const __bf16* __restrict__ Z,
                     const float* __restrict__ W,     // [1024][1024]
                     const float* __restrict__ bias,  // [1024]
                     float* __restrict__ Out) {
  __shared__ __bf16 As[128][32 + 8];
  __shared__ __bf16 BsT[64][32 + 8];

  const int tid  = threadIdx.x;
  const int wid  = tid >> 5;
  const int lane = tid & 31;
  const int r0   = blockIdx.x * 128;
  const int c0   = blockIdx.y * 64;

  v8f acc[4];
#pragma unroll
  for (int i = 0; i < 4; ++i) acc[i] = zero8();

  const int arow = wid * 16 + (lane & 15);
  const int kbA  = (lane < 16) ? 0 : 8;
  const int kbB  = (lane < 16) ? 0 : 8;

  for (int kk = 0; kk < D_MODEL; kk += 32) {
    __syncthreads();
#pragma unroll
    for (int i = 0; i < 2; ++i) {             // A tile: pure b128 copies
      int e = tid + i * 256;                  // 512 groups of 8
      int row = e >> 2, c8 = (e & 3) * 8;
      *(v8bf*)&As[row][c8] =
          *(const v8bf*)&Z[(size_t)(r0 + row) * D_MODEL + kk + c8];
    }
#pragma unroll
    for (int i = 0; i < 4; ++i) {             // W_O tile, transposed pairs
      int e = tid + i * 256;
      int col = e & 63, krp = (e >> 6) * 2;
      size_t wb = (size_t)(kk + krp) * D_MODEL + c0 + col;
      *(v2bf*)&BsT[col][krp] = pack2(W[wb], W[wb + D_MODEL]);
    }
    __syncthreads();

    v16bf a = frag_cat(&As[arow][kbA], &As[arow][kbA + 16]);
#pragma unroll
    for (int bn = 0; bn < 4; ++bn) {
      int col = bn * 16 + (lane & 15);
      v16bf bfrag = frag_cat(&BsT[col][kbB], &BsT[col][kbB] + 16);
      acc[bn] = wmma_bf16(a, bfrag, acc[bn]);
    }
  }

  const int mof = (lane < 16) ? 0 : 8;
#pragma unroll
  for (int bn = 0; bn < 4; ++bn) {
    int col = c0 + bn * 16 + (lane & 15);
    float bv = bias[col];
#pragma unroll
    for (int i = 0; i < 8; ++i) {
      int row = r0 + wid * 16 + i + mof;
      Out[(size_t)row * D_MODEL + col] = acc[bn][i] + bv;
    }
  }
}

// =====================================================================
// Launcher
// =====================================================================
extern "C" void kernel_launch(void* const* d_in, const int* in_sizes, int n_in,
                              void* d_out, int out_size, void* d_ws, size_t ws_size,
                              hipStream_t stream) {
  const float*   x_q  = (const float*)d_in[0];
  const float*   x_kv = (const float*)d_in[1];
  const uint8_t* mask = (const uint8_t*)d_in[2];   // numpy bool = 1 byte
  const float*   W_Q  = (const float*)d_in[3];
  const float*   W_K  = (const float*)d_in[4];
  const float*   W_V  = (const float*)d_in[5];
  const float*   W_O  = (const float*)d_in[6];
  const float*   b_Q  = (const float*)d_in[7];
  const float*   b_K  = (const float*)d_in[8];
  const float*   b_V  = (const float*)d_in[9];
  const float*   b_O  = (const float*)d_in[10];

  float* out  = (float*)d_out;                      // [8192][1024]
  float* zout = out + (size_t)MROWS * D_MODEL;      // z, bsnh flattened

  // workspace: bf16 q | k | v | z (16 MB each)
  __bf16* q = (__bf16*)d_ws;
  __bf16* k = q + (size_t)MROWS * D_MODEL;
  __bf16* v = k + (size_t)MROWS * D_MODEL;
  __bf16* z = v + (size_t)MROWS * D_MODEL;

  dim3 blk(256);
  dim3 gProj(MROWS / 128, D_MODEL / 64);            // (64, 16)
  qkv_proj_kernel<<<gProj, blk, 0, stream>>>(x_q,  W_Q, b_Q, q);
  qkv_proj_kernel<<<gProj, blk, 0, stream>>>(x_kv, W_K, b_K, k);
  qkv_proj_kernel<<<gProj, blk, 0, stream>>>(x_kv, W_V, b_V, v);

  dim3 gAttn(SEQ / 128, N_HEADS, BATCH);            // (16, 16, 4)
  attn_kernel<<<gAttn, blk, 0, stream>>>(q, k, v, mask, zout, z);

  out_proj_kernel<<<gProj, blk, 0, stream>>>(z, W_O, b_O, out);
}
